// GLFAM_87385404604508
// MI455X (gfx1250) — matrix operand order repaired and checked by reading, not compiled, not run on hardware
//
#include <hip/hip_runtime.h>
#include <hip/hip_bf16.h>

typedef __attribute__((ext_vector_type(16))) _Float16 v16h;
typedef __attribute__((ext_vector_type(8)))  _Float16 v8h;
typedef __attribute__((ext_vector_type(8)))  float    v8f;

#define WMMA_F16(a, b, c) \
    __builtin_amdgcn_wmma_f32_16x16x32_f16(false, (a), false, (b), (short)0, (c), false, false)

__device__ __forceinline__ float gelu_exact(float v) {
    return 0.5f * v * (1.0f + erff(v * 0.70710678118654752f));
}

// A-matrix (16x32 f16) fragment: lane L holds row M=L%16.
// element e -> K = kb + e + (e>=8 ? 8 : 0), where kb = 8*(L>=16). Two 16B runs.
__device__ __forceinline__ v16h load_a_frag(const _Float16* rowbase, int kb) {
    v8h lo = *(const v8h*)(rowbase + kb);
    v8h hi = *(const v8h*)(rowbase + kb + 16);
    v16h a;
#pragma unroll
    for (int e = 0; e < 8; ++e) { a[e] = lo[e]; a[e + 8] = hi[e]; }
    return a;
}

// B-matrix (32x16 f16) fragment: lane L holds col N=L%16;
// element e -> K = kb2/2 + e + (e>=8 ? 8 : 0), kb2 = 16*(L>=16). Two 16B runs.
__device__ __forceinline__ v16h load_b_frag(const _Float16* colbase, int kb2) {
    v8h lo = *(const v8h*)(colbase + kb2 / 2);
    v8h hi = *(const v8h*)(colbase + kb2 / 2 + 16);
    v16h b;
#pragma unroll
    for (int e = 0; e < 8; ++e) { b[e] = lo[e]; b[e + 8] = hi[e]; }
    return b;
}

__device__ __forceinline__ float rowmax16(float v) {
#pragma unroll
    for (int m = 1; m < 16; m <<= 1) v = fmaxf(v, __shfl_xor(v, m, 16));
    return v;
}
__device__ __forceinline__ float rowsum16(float v) {
#pragma unroll
    for (int m = 1; m < 16; m <<= 1) v += __shfl_xor(v, m, 16);
    return v;
}

// ---------------------------------------------------------------- constants
// B=8, C=128, H=W=64, N=4096. x1 = ch 0..63, x2 = ch 64..127.

// ------------------------------------------------ K0: w_def -> f16 [64][576]
__global__ void wdef_cast(const float* __restrict__ w_def, _Float16* __restrict__ Bw16) {
    int i = blockIdx.x * 256 + threadIdx.x;
    if (i < 64 * 576) Bw16[i] = (_Float16)w_def[i];
}

// ------------------------------------------------ K1: offset conv (64 -> 18)
__global__ void offset_conv(const float* __restrict__ x, const float* __restrict__ w_off,
                            const float* __restrict__ b_off, float* __restrict__ off) {
    int idx = blockIdx.x * 256 + threadIdx.x;     // < 8*18*4096
    int p = idx & 4095;
    int rest = idx >> 12;
    int oc = rest % 18;
    int b = rest / 18;
    int h = p >> 6, w = p & 63;
    const float* xb = x + (size_t)b * 128 * 4096;
    float acc = b_off[oc];
#pragma unroll
    for (int tap = 0; tap < 9; ++tap) {
        int hh = h + tap / 3 - 1, ww = w + tap % 3 - 1;
        if ((unsigned)hh < 64u && (unsigned)ww < 64u) {
            int pp = hh * 64 + ww;
            const float* wrow = w_off + (size_t)oc * 576 + tap;
            for (int c = 0; c < 64; ++c)
                acc += xb[(size_t)c * 4096 + pp] * wrow[(size_t)c * 9];
        }
    }
    off[idx] = acc;
}

// --------------------------------- K2: 1x1 qkv conv + GELU, route q/k/v/z1
__global__ void qkv_kernel(const float* __restrict__ x, const float* __restrict__ w_qkv,
                           const float* __restrict__ b_qkv, _Float16* __restrict__ q16,
                           _Float16* __restrict__ k16, _Float16* __restrict__ v16,
                           float* __restrict__ out) {
    int idx = blockIdx.x * 256 + threadIdx.x;     // < 8*128*4096
    int p = idx & 4095;
    int rest = idx >> 12;
    int co = rest & 127;
    int b = rest >> 7;
    const float* xb = x + ((size_t)b * 128 + 64) * 4096 + p;
    const float* wr = w_qkv + (size_t)co * 64;
    float acc = b_qkv[co];
#pragma unroll 8
    for (int ci = 0; ci < 64; ++ci) acc += xb[(size_t)ci * 4096] * wr[ci];
    acc = gelu_exact(acc);
    if (co < 32)       q16[((size_t)b * 4096 + p) * 32 + co] = (_Float16)acc;
    else if (co < 64)  k16[((size_t)b * 4096 + p) * 32 + (co - 32)] = (_Float16)acc;
    else if (co < 96)  v16[((size_t)b * 32 + (co - 64)) * 4096 + p] = (_Float16)acc;
    else               out[((size_t)b * 128 + 64 + (co - 96)) * 4096 + p] = acc;  // z1
}

// --------------- K3: deformable sampling + WMMA GEMM (M=16 pix, N=64, K=576)
__global__ void deform_kernel(const float* __restrict__ x, const float* __restrict__ off,
                              const _Float16* __restrict__ Bw16,
                              const float* __restrict__ b_def, float* __restrict__ out) {
    __shared__ __align__(16) _Float16 As[16][576];
    __shared__ float sy[16][9], sx[16][9];

    int b = blockIdx.x >> 8;
    int pixbase = (blockIdx.x & 255) * 16;
    int t = threadIdx.x;

    if (t < 144) {   // per-(pixel,tap) sample coordinates
        int pix = t / 9, tap = t % 9;
        int p = pixbase + pix;
        int hh = p >> 6, ww = p & 63;
        float oy = off[(((size_t)b * 9 + tap) * 2 + 0) * 4096 + p];
        float ox = off[(((size_t)b * 9 + tap) * 2 + 1) * 4096 + p];
        sy[pix][tap] = (float)(hh + tap / 3 - 1) + oy;
        sx[pix][tap] = (float)(ww + tap % 3 - 1) + ox;
    }
    __syncthreads();

    const float* xb = x + (size_t)b * 128 * 4096;     // x1 = first 64 channels
    for (int s = t; s < 16 * 576; s += 256) {
        int pix = s / 576;
        int kidx = s % 576;       // kidx = c*9 + tap  (matches w_def reshape)
        int c = kidx / 9, tap = kidx % 9;
        float ys = sy[pix][tap], xs = sx[pix][tap];
        float fy = floorf(ys), fx = floorf(xs);
        int y0 = (int)fy, x0 = (int)fx;
        float wy1 = ys - fy, wx1 = xs - fx;
        float wy0 = 1.0f - wy1, wx0 = 1.0f - wx1;
        const float* ch = xb + (size_t)c * 4096;
        float v00 = ((unsigned)y0 < 64u && (unsigned)x0 < 64u) ? ch[y0 * 64 + x0] : 0.0f;
        float v01 = ((unsigned)y0 < 64u && (unsigned)(x0 + 1) < 64u) ? ch[y0 * 64 + x0 + 1] : 0.0f;
        float v10 = ((unsigned)(y0 + 1) < 64u && (unsigned)x0 < 64u) ? ch[(y0 + 1) * 64 + x0] : 0.0f;
        float v11 = ((unsigned)(y0 + 1) < 64u && (unsigned)(x0 + 1) < 64u) ? ch[(y0 + 1) * 64 + x0 + 1] : 0.0f;
        As[pix][kidx] = (_Float16)(wy0 * (wx0 * v00 + wx1 * v01) + wy1 * (wx0 * v10 + wx1 * v11));
    }
    __syncthreads();

    int wave = t >> 5, lane = t & 31;
    int lane16 = lane & 15, hi = lane >> 4;
    if (wave < 4) {                       // 4 N-tiles of 16 output channels
        v8f acc = {};
        const _Float16* brow = Bw16 + (size_t)(wave * 16 + lane16) * 576;
        const _Float16* arow = &As[lane16][0];
#pragma unroll
        for (int step = 0; step < 18; ++step) {
            v16h a = load_a_frag(arow + step * 32, hi * 8);
            v16h bb = load_b_frag(brow + step * 32, hi * 16);
            acc = WMMA_F16(a, bb, acc);
        }
        int o = wave * 16 + lane16;
        float bias = b_def[o];
        float* op = out + ((size_t)b * 128 + o) * 4096 + pixbase;
#pragma unroll
        for (int r = 0; r < 8; ++r)
            op[r + hi * 8] = gelu_exact(acc[r] + bias);
    }
}

// ---------------------- K4: flash attention, d=32, N=4096, f16 WMMA + f32 acc
__global__ void flash_attn(const _Float16* __restrict__ q16, const _Float16* __restrict__ k16,
                           const _Float16* __restrict__ v16, float* __restrict__ out) {
    __shared__ __align__(16) _Float16 Psh[8][16][32];

    int b = blockIdx.x >> 5;                 // 8 batches
    int wave = threadIdx.x >> 5;             // 8 waves, one 16-row i-tile each
    int lane = threadIdx.x & 31;
    int lane16 = lane & 15, hi = lane >> 4;
    int i_base = ((blockIdx.x & 31) * 8 + wave) * 16;

    const float scale = 0.08838834764831845f;   // 1/sqrt(128)

    // Q fragment, rows = queries, K-dim = head dim (32)
    const _Float16* qrow = q16 + ((size_t)b * 4096 + i_base + lane16) * 32;
    v16h a_q = load_a_frag(qrow, hi * 8);

    const _Float16* kbase = k16 + (size_t)b * 4096 * 32;
    const _Float16* vbase = v16 + (size_t)b * 32 * 4096;

    float mrow[8], lrow[8];
#pragma unroll
    for (int r = 0; r < 8; ++r) { mrow[r] = -1.0e30f; lrow[r] = 0.0f; }
    v8f acc0 = {}, acc1 = {};

    for (int jb = 0; jb < 4096; jb += 32) {
        // S = Q^T K for two 16-column groups
        v16h bk0 = load_b_frag(kbase + (size_t)(jb + lane16) * 32, hi * 16);
        v16h bk1 = load_b_frag(kbase + (size_t)(jb + 16 + lane16) * 32, hi * 16);
        v8f z0 = {}, z1 = {};
        v8f s0 = WMMA_F16(a_q, bk0, z0);
        v8f s1 = WMMA_F16(a_q, bk1, z1);

        float p0[8], p1[8];
#pragma unroll
        for (int r = 0; r < 8; ++r) {      // row = r + 8*hi, col = lane16
            float sv0 = s0[r] * scale, sv1 = s1[r] * scale;
            float rm = rowmax16(fmaxf(sv0, sv1));
            float mnew = fmaxf(mrow[r], rm);
            float alpha = __expf(mrow[r] - mnew);
            float e0 = __expf(sv0 - mnew);
            float e1 = __expf(sv1 - mnew);
            lrow[r] = lrow[r] * alpha + rowsum16(e0 + e1);
            mrow[r] = mnew;
            acc0[r] *= alpha;
            acc1[r] *= alpha;
            p0[r] = e0;
            p1[r] = e1;
        }

        // repack P (C-tile layout) -> A-fragment layout through LDS
#pragma unroll
        for (int r = 0; r < 8; ++r) {
            Psh[wave][r + hi * 8][lane16] = (_Float16)p0[r];
            Psh[wave][r + hi * 8][lane16 + 16] = (_Float16)p1[r];
        }
        __syncthreads();
        v16h a_p = load_a_frag(&Psh[wave][lane16][0], hi * 8);

        // acc += P * V^T  (K-dim = 32 j's, N = head dims 0..15 / 16..31)
        v16h bv0 = load_b_frag(vbase + (size_t)lane16 * 4096 + jb, hi * 16);
        v16h bv1 = load_b_frag(vbase + (size_t)(16 + lane16) * 4096 + jb, hi * 16);
        acc0 = WMMA_F16(a_p, bv0, acc0);
        acc1 = WMMA_F16(a_p, bv1, acc1);
    }

    // out2 -> output channels 96..127
    float* op = out + ((size_t)b * 128 + 96) * 4096;
#pragma unroll
    for (int r = 0; r < 8; ++r) {
        float inv = 1.0f / lrow[r];
        int i = i_base + r + hi * 8;
        op[(size_t)lane16 * 4096 + i] = acc0[r] * inv;
        op[(size_t)(16 + lane16) * 4096 + i] = acc1[r] * inv;
    }
}

// --------------------------------------------------------------------------
extern "C" void kernel_launch(void* const* d_in, const int* in_sizes, int n_in,
                              void* d_out, int out_size, void* d_ws, size_t ws_size,
                              hipStream_t stream) {
    (void)in_sizes; (void)n_in; (void)out_size; (void)ws_size;
    const float* x     = (const float*)d_in[0];
    const float* w_off = (const float*)d_in[1];
    const float* b_off = (const float*)d_in[2];
    const float* w_def = (const float*)d_in[3];
    const float* b_def = (const float*)d_in[4];
    const float* w_qkv = (const float*)d_in[5];
    const float* b_qkv = (const float*)d_in[6];
    float* out = (float*)d_out;

    char* ws = (char*)d_ws;
    float*    off  = (float*)ws;                       // 8*18*4096 f32 = 2,359,296 B
    _Float16* Bw16 = (_Float16*)(ws + 2359296);        // 64*576 f16   =    73,728 B
    _Float16* q16  = (_Float16*)(ws + 2433024);        // 8*4096*32 f16 = 2,097,152 B
    _Float16* k16  = (_Float16*)(ws + 4530176);        // 2,097,152 B
    _Float16* v16  = (_Float16*)(ws + 6627328);        // 2,097,152 B  (total ~8.7 MB)

    wdef_cast   <<<144,   256, 0, stream>>>(w_def, Bw16);
    offset_conv <<<2304,  256, 0, stream>>>(x, w_off, b_off, off);
    qkv_kernel  <<<16384, 256, 0, stream>>>(x, w_qkv, b_qkv, q16, k16, v16, out);
    deform_kernel<<<2048, 256, 0, stream>>>(x, off, Bw16, b_def, out);
    flash_attn  <<<256,   256, 0, stream>>>(q16, k16, v16, out);
}
